// AttentionLayer_9552007266944
// MI455X (gfx1250) — compile-verified, matrix operand order checked by reading, not run on hardware
//
#include <hip/hip_runtime.h>

// ---------------------------------------------------------------------------
// Problem dims (fixed by the reference)
#define T_DIM 16384
#define J_DIM 1024
#define D2    512

typedef __bf16 bf16_t;
typedef __attribute__((ext_vector_type(8)))  bf16_t v8bf;
typedef __attribute__((ext_vector_type(16))) bf16_t v16bf;
typedef __attribute__((ext_vector_type(8)))  float  v8f;

// ---------------------------------------------------------------------------
// Helpers

__device__ __forceinline__ unsigned short f32_to_bf16_bits(float f) {
  unsigned int u = __float_as_uint(f);
  u += 0x7FFFu + ((u >> 16) & 1u);   // round-to-nearest-even
  return (unsigned short)(u >> 16);
}

__device__ __forceinline__ float wave_reduce_sum(float v) {
  #pragma unroll
  for (int m = 16; m >= 1; m >>= 1) v += __shfl_xor(v, m, 32);
  return v;
}

__device__ __forceinline__ float wave_reduce_max(float v) {
  #pragma unroll
  for (int m = 16; m >= 1; m >>= 1) v = fmaxf(v, __shfl_xor(v, m, 32));
  return v;
}

// Load a 16x32 bf16 WMMA A/B fragment from an N-major (row-major over the
// 16-dim) matrix. Per ISA 16-bit A layout: lane = m + 16*half,
// VGPR0..3 hold K = half*8 .. half*8+7, VGPR4..7 hold K = 16+half*8 .. +7.
__device__ __forceinline__ v16bf load_frag(const unsigned short* __restrict__ rowptr,
                                           int k0, int half) {
  v8bf lo = *(const v8bf*)(rowptr + k0 + half * 8);
  v8bf hi = *(const v8bf*)(rowptr + k0 + 16 + half * 8);
  return __builtin_shufflevector(lo, hi, 0, 1, 2, 3, 4, 5, 6, 7,
                                         8, 9, 10, 11, 12, 13, 14, 15);
}

__device__ __forceinline__ v8f wmma_bf16(v16bf a, v16bf b, v8f c) {
  return __builtin_amdgcn_wmma_f32_16x16x32_bf16(false, a, false, b,
                                                 (short)0, c, false, false);
}

// ---------------------------------------------------------------------------
// K1: per-row of H: hw1[t] = H[t,:].w1 ; A[t,:] = bf16(H[t,:]*w3)
__global__ void prep_H_kernel(const float* __restrict__ H,
                              const float* __restrict__ Ws,
                              unsigned short* __restrict__ A,
                              float* __restrict__ hw1) {
  int t    = blockIdx.x * (blockDim.x >> 5) + (threadIdx.x >> 5);
  int lane = threadIdx.x & 31;
  const float* __restrict__ w1 = Ws;            // Ws[0:512]
  const float* __restrict__ w3 = Ws + 2 * D2;   // Ws[1024:1536]
  const float* __restrict__ h  = H + (size_t)t * D2;
  unsigned short* __restrict__ a = A + (size_t)t * D2;
  float acc = 0.0f;
  #pragma unroll
  for (int i = 0; i < D2 / 32; ++i) {
    int d = lane + i * 32;
    float hv = h[d];
    acc += hv * w1[d];
    a[d] = f32_to_bf16_bits(hv * w3[d]);
  }
  acc = wave_reduce_sum(acc);
  if (lane == 0) hw1[t] = acc;
}

// K2: per-row of U: uw2[j] = U[j,:].w2 ; Ub = bf16(U) ; UbT = bf16(U)^T
__global__ void prep_U_kernel(const float* __restrict__ U,
                              const float* __restrict__ Ws,
                              unsigned short* __restrict__ Ub,
                              unsigned short* __restrict__ UbT,
                              float* __restrict__ uw2) {
  int j    = blockIdx.x * (blockDim.x >> 5) + (threadIdx.x >> 5);
  int lane = threadIdx.x & 31;
  const float* __restrict__ w2 = Ws + D2;       // Ws[512:1024]
  const float* __restrict__ u  = U + (size_t)j * D2;
  float acc = 0.0f;
  #pragma unroll
  for (int i = 0; i < D2 / 32; ++i) {
    int d = lane + i * 32;
    float uv = u[d];
    acc += uv * w2[d];
    unsigned short bits = f32_to_bf16_bits(uv);
    Ub [(size_t)j * D2 + d]    = bits;
    UbT[(size_t)d * J_DIM + j] = bits;
  }
  acc = wave_reduce_sum(acc);
  if (lane == 0) uw2[j] = acc;
}

// K3: S = A @ Ub^T + hw1[:,None] + uw2[None,:]
// Register-blocked: each wave computes a 32x64 tile (2 M-frags x 4 N-frags):
// per k-step 12 b128 loads feed 8 WMMAs.
__global__ void gemm_S_kernel(const unsigned short* __restrict__ A,
                              const unsigned short* __restrict__ Ub,
                              const float* __restrict__ hw1,
                              const float* __restrict__ uw2,
                              float* __restrict__ S) {
  int tile = blockIdx.x * (blockDim.x >> 5) + (threadIdx.x >> 5);
  int lane = threadIdx.x & 31;
  int m    = lane & 15;
  int half = lane >> 4;
  int tileN = tile & (J_DIM / 64 - 1);   // 16 tiles across
  int tileM = tile >> 4;                 // 512 down
  int row0 = tileM * 32, col0 = tileN * 64;

  const unsigned short* __restrict__ arow0 = A + (size_t)(row0 + m) * D2;
  const unsigned short* __restrict__ arow1 = A + (size_t)(row0 + 16 + m) * D2;
  const unsigned short* __restrict__ brow0 = Ub + (size_t)(col0 + m) * D2;
  const unsigned short* __restrict__ brow1 = Ub + (size_t)(col0 + 16 + m) * D2;
  const unsigned short* __restrict__ brow2 = Ub + (size_t)(col0 + 32 + m) * D2;
  const unsigned short* __restrict__ brow3 = Ub + (size_t)(col0 + 48 + m) * D2;

  v8f c00 = {}, c01 = {}, c02 = {}, c03 = {};
  v8f c10 = {}, c11 = {}, c12 = {}, c13 = {};

  #pragma unroll 2
  for (int k0 = 0; k0 < D2; k0 += 32) {
    if (k0 + 32 < D2) {   // uniform (scalar) branch: EXEC untouched
      __builtin_prefetch(arow0 + k0 + 32, 0, 1);
      __builtin_prefetch(arow1 + k0 + 32, 0, 1);
    }
    v16bf a0 = load_frag(arow0, k0, half);
    v16bf a1 = load_frag(arow1, k0, half);
    v16bf b0 = load_frag(brow0, k0, half);
    v16bf b1 = load_frag(brow1, k0, half);
    v16bf b2 = load_frag(brow2, k0, half);
    v16bf b3 = load_frag(brow3, k0, half);
    c00 = wmma_bf16(a0, b0, c00);
    c10 = wmma_bf16(a1, b0, c10);
    c01 = wmma_bf16(a0, b1, c01);
    c11 = wmma_bf16(a1, b1, c11);
    c02 = wmma_bf16(a0, b2, c02);
    c12 = wmma_bf16(a1, b2, c12);
    c03 = wmma_bf16(a0, b3, c03);
    c13 = wmma_bf16(a1, b3, c13);
  }

  int n = lane & 15;
  v8f acc[2][4] = {{c00, c01, c02, c03}, {c10, c11, c12, c13}};
  #pragma unroll
  for (int mi = 0; mi < 2; ++mi) {
    #pragma unroll
    for (int ni = 0; ni < 4; ++ni) {
      float un = uw2[col0 + ni * 16 + n];
      #pragma unroll
      for (int r = 0; r < 8; ++r) {
        int row = row0 + mi * 16 + r + half * 8;
        S[(size_t)row * J_DIM + col0 + ni * 16 + n] =
            acc[mi][ni][r] + hw1[row] + un;
      }
    }
  }
}

// K4: per-row softmax of S -> P (bf16, normalized) ; also rowmax m[t]
__global__ void softmax_rows_kernel(const float* __restrict__ S,
                                    unsigned short* __restrict__ P,
                                    float* __restrict__ mrow) {
  int t    = blockIdx.x * (blockDim.x >> 5) + (threadIdx.x >> 5);
  int lane = threadIdx.x & 31;
  const float* __restrict__ srow = S + (size_t)t * J_DIM;
  float v[J_DIM / 32];
  float mx = -__builtin_inff();
  #pragma unroll
  for (int i = 0; i < J_DIM / 32; ++i) {
    v[i] = srow[lane + i * 32];
    mx = fmaxf(mx, v[i]);
  }
  mx = wave_reduce_max(mx);
  float sum = 0.0f;
  #pragma unroll
  for (int i = 0; i < J_DIM / 32; ++i) {
    v[i] = __expf(v[i] - mx);
    sum += v[i];
  }
  sum = wave_reduce_sum(sum);
  float inv = 1.0f / sum;
  unsigned short* __restrict__ prow = P + (size_t)t * J_DIM;
  #pragma unroll
  for (int i = 0; i < J_DIM / 32; ++i)
    prow[lane + i * 32] = f32_to_bf16_bits(v[i] * inv);
  if (lane == 0) mrow[t] = mx;
}

// K5: b = softmax over T of rowmax   (single 1024-thread block)
__global__ void softmax_b_kernel(const float* __restrict__ mrow,
                                 float* __restrict__ bvec) {
  __shared__ float red[1024];
  int tid = threadIdx.x;
  float mx = -__builtin_inff();
  #pragma unroll
  for (int i = 0; i < T_DIM / 1024; ++i) mx = fmaxf(mx, mrow[tid + i * 1024]);
  red[tid] = mx;
  __syncthreads();
  for (int s = 512; s > 0; s >>= 1) {
    if (tid < s) red[tid] = fmaxf(red[tid], red[tid + s]);
    __syncthreads();
  }
  float gmax = red[0];
  __syncthreads();
  float sum = 0.0f;
  #pragma unroll
  for (int i = 0; i < T_DIM / 1024; ++i) sum += __expf(mrow[tid + i * 1024] - gmax);
  red[tid] = sum;
  __syncthreads();
  for (int s = 512; s > 0; s >>= 1) {
    if (tid < s) red[tid] += red[tid + s];
    __syncthreads();
  }
  float inv = 1.0f / red[0];
  #pragma unroll
  for (int i = 0; i < T_DIM / 1024; ++i) {
    int t = tid + i * 1024;
    bvec[t] = __expf(mrow[t] - gmax) * inv;
  }
}

// K6/K7: h_tilde = b @ H  (partials over 16 chunks of T, then combine)
__global__ void htilde_part_kernel(const float* __restrict__ H,
                                   const float* __restrict__ bvec,
                                   float* __restrict__ part) {
  int d = threadIdx.x;        // 512 threads
  int g = blockIdx.x;         // 16 chunks
  int t0 = g * (T_DIM / 16);
  float acc = 0.0f;
  for (int i = 0; i < T_DIM / 16; ++i)
    acc = fmaf(bvec[t0 + i], H[(size_t)(t0 + i) * D2 + d], acc);
  part[g * D2 + d] = acc;
}

__global__ void htilde_final_kernel(const float* __restrict__ part,
                                    float* __restrict__ ht) {
  int d = threadIdx.x;
  float acc = 0.0f;
  #pragma unroll
  for (int g = 0; g < 16; ++g) acc += part[g * D2 + d];
  ht[d] = acc;
}

// K8: attended = P @ U (WMMA, 32x64 register-blocked), fused with G assembly:
// G = [H | attended | H*attended | H*h_tilde]
__global__ void gemm_att_out_kernel(const unsigned short* __restrict__ P,
                                    const unsigned short* __restrict__ UbT,
                                    const float* __restrict__ H,
                                    const float* __restrict__ ht,
                                    float* __restrict__ G) {
  int tile = blockIdx.x * (blockDim.x >> 5) + (threadIdx.x >> 5);
  int lane = threadIdx.x & 31;
  int m    = lane & 15;
  int half = lane >> 4;
  int tileN = tile & (D2 / 64 - 1);   // 8 tiles across
  int tileM = tile >> 3;              // 512 down
  int row0 = tileM * 32, col0 = tileN * 64;

  const unsigned short* __restrict__ arow0 = P + (size_t)(row0 + m) * J_DIM;
  const unsigned short* __restrict__ arow1 = P + (size_t)(row0 + 16 + m) * J_DIM;
  const unsigned short* __restrict__ brow0 = UbT + (size_t)(col0 + m) * J_DIM;
  const unsigned short* __restrict__ brow1 = UbT + (size_t)(col0 + 16 + m) * J_DIM;
  const unsigned short* __restrict__ brow2 = UbT + (size_t)(col0 + 32 + m) * J_DIM;
  const unsigned short* __restrict__ brow3 = UbT + (size_t)(col0 + 48 + m) * J_DIM;

  v8f c00 = {}, c01 = {}, c02 = {}, c03 = {};
  v8f c10 = {}, c11 = {}, c12 = {}, c13 = {};

  #pragma unroll 2
  for (int k0 = 0; k0 < J_DIM; k0 += 32) {
    if (k0 + 32 < J_DIM) {
      __builtin_prefetch(arow0 + k0 + 32, 0, 1);
      __builtin_prefetch(arow1 + k0 + 32, 0, 1);
    }
    v16bf a0 = load_frag(arow0, k0, half);
    v16bf a1 = load_frag(arow1, k0, half);
    v16bf b0 = load_frag(brow0, k0, half);
    v16bf b1 = load_frag(brow1, k0, half);
    v16bf b2 = load_frag(brow2, k0, half);
    v16bf b3 = load_frag(brow3, k0, half);
    c00 = wmma_bf16(a0, b0, c00);
    c10 = wmma_bf16(a1, b0, c10);
    c01 = wmma_bf16(a0, b1, c01);
    c11 = wmma_bf16(a1, b1, c11);
    c02 = wmma_bf16(a0, b2, c02);
    c12 = wmma_bf16(a1, b2, c12);
    c03 = wmma_bf16(a0, b3, c03);
    c13 = wmma_bf16(a1, b3, c13);
  }

  v8f acc[2][4] = {{c00, c01, c02, c03}, {c10, c11, c12, c13}};
  #pragma unroll
  for (int ni = 0; ni < 4; ++ni) {
    int n = col0 + ni * 16 + (lane & 15);
    float htn = ht[n];
    #pragma unroll
    for (int mi = 0; mi < 2; ++mi) {
      #pragma unroll
      for (int r = 0; r < 8; ++r) {
        int t = row0 + mi * 16 + r + half * 8;
        float h   = H[(size_t)t * D2 + n];
        float att = acc[mi][ni][r];
        float* __restrict__ grow = G + (size_t)t * (4 * D2);
        grow[n]          = h;
        grow[D2 + n]     = att;
        grow[2 * D2 + n] = h * att;
        grow[3 * D2 + n] = h * htn;
      }
    }
  }
}

// ---------------------------------------------------------------------------

extern "C" void kernel_launch(void* const* d_in, const int* in_sizes, int n_in,
                              void* d_out, int out_size, void* d_ws, size_t ws_size,
                              hipStream_t stream) {
  (void)in_sizes; (void)n_in; (void)out_size; (void)ws_size;

  const float* H  = (const float*)d_in[0];
  const float* U  = (const float*)d_in[1];
  const float* Ws = (const float*)d_in[2];
  float* G = (float*)d_out;

  // Workspace carve-up (256-byte aligned slices)
  char* ws = (char*)d_ws;
  size_t off = 0;
  auto carve = [&](size_t bytes) -> char* {
    char* p = ws + off;
    off = (off + bytes + 255) & ~(size_t)255;
    return p;
  };
  unsigned short* A    = (unsigned short*)carve((size_t)T_DIM * D2 * 2);   // 16 MB
  unsigned short* Ub   = (unsigned short*)carve((size_t)J_DIM * D2 * 2);   //  1 MB
  unsigned short* UbT  = (unsigned short*)carve((size_t)D2 * J_DIM * 2);   //  1 MB
  float*          hw1  = (float*)carve((size_t)T_DIM * 4);
  float*          uw2  = (float*)carve((size_t)J_DIM * 4);
  float*          S    = (float*)carve((size_t)T_DIM * J_DIM * 4);         // 64 MB
  unsigned short* P    = (unsigned short*)carve((size_t)T_DIM * J_DIM * 2);// 32 MB
  float*          mrow = (float*)carve((size_t)T_DIM * 4);
  float*          bvec = (float*)carve((size_t)T_DIM * 4);
  float*          part = (float*)carve((size_t)16 * D2 * 4);
  float*          ht   = (float*)carve((size_t)D2 * 4);

  // 1) prep H: 16384 waves, 8 waves/block
  prep_H_kernel<<<T_DIM / 8, 256, 0, stream>>>(H, Ws, A, hw1);
  // 2) prep U: 1024 waves
  prep_U_kernel<<<J_DIM / 8, 256, 0, stream>>>(U, Ws, Ub, UbT, uw2);
  // 3) S GEMM: (T/32)*(J/64) = 8192 wave-tiles, 8/block
  gemm_S_kernel<<<(T_DIM / 32) * (J_DIM / 64) / 8, 256, 0, stream>>>(A, Ub, hw1, uw2, S);
  // 4) row softmax: one wave per row
  softmax_rows_kernel<<<T_DIM / 8, 256, 0, stream>>>(S, P, mrow);
  // 5) b = softmax(rowmax): single block
  softmax_b_kernel<<<1, 1024, 0, stream>>>(mrow, bvec);
  // 6/7) h_tilde
  htilde_part_kernel<<<16, D2, 0, stream>>>(H, bvec, part);
  htilde_final_kernel<<<1, D2, 0, stream>>>(part, ht);
  // 8) attended GEMM + G assembly: (T/32)*(D2/64) = 4096 wave-tiles
  gemm_att_out_kernel<<<(T_DIM / 32) * (D2 / 64) / 8, 256, 0, stream>>>(P, UbT, H, ht, G);
}
